// MultiRouteRosaLayer_32796370272645
// MI455X (gfx1250) — matrix operand order repaired and checked by reading, not compiled, not run on hardware
//
#include <hip/hip_runtime.h>
#include <hip/hip_bf16.h>
#include <stdint.h>

// Problem constants (match reference)
#define M_    2
#define B_    4
#define T_    2048
#define D_    1024
#define KQK_  512
#define VV_   8192
#define TOPK_ 8
#define ROWS_ (B_ * T_)      // 8192 rows of x
#define NST_  (2 * T_ + 10)  // max SAM states = 4106

// ---- CDNA5 WMMA types ----
typedef __bf16 v16bf __attribute__((ext_vector_type(16)));
typedef float  v8f   __attribute__((ext_vector_type(8)));

union ABFrag { uint4 q[2]; v16bf v; };

__device__ __forceinline__ unsigned short f2bf(float f) {
  unsigned u = __float_as_uint(f);
  u += 0x7FFFu + ((u >> 16) & 1u);   // round-to-nearest-even
  return (unsigned short)(u >> 16);
}
// map float to monotonically ordered unsigned
__device__ __forceinline__ unsigned f2ord(float f) {
  unsigned u = __float_as_uint(f);
  return (u & 0x80000000u) ? ~u : (u | 0x80000000u);
}
__device__ __forceinline__ unsigned long long sxor64(unsigned long long v, int m) {
  unsigned lo = __shfl_xor((unsigned)v, m, 32);
  unsigned hi = __shfl_xor((unsigned)(v >> 32), m, 32);
  return ((unsigned long long)hi << 32) | lo;
}

// ---------------- utility kernels ----------------
__global__ void k_fill_u32(unsigned* __restrict__ p, size_t n, unsigned v) {
  size_t i = (size_t)blockIdx.x * blockDim.x + threadIdx.x;
  size_t st = (size_t)gridDim.x * blockDim.x;
  for (; i < n; i += st) p[i] = v;
}

__global__ void k_cvt_bf16(const float* __restrict__ s, unsigned short* __restrict__ d, size_t n4) {
  size_t i = (size_t)blockIdx.x * blockDim.x + threadIdx.x;
  size_t st = (size_t)gridDim.x * blockDim.x;
  for (; i < n4; i += st) {
    float4 f = ((const float4*)s)[i];
    uint2 o;
    o.x = (unsigned)f2bf(f.x) | ((unsigned)f2bf(f.y) << 16);
    o.y = (unsigned)f2bf(f.z) | ((unsigned)f2bf(f.w) << 16);
    ((uint2*)d)[i] = o;
  }
}

// ---------------- WMMA GEMM + fused argmax ----------------
// scores = Xb (ROWS_ x D_) * W(m)^T (D_ x N); per-row argmax -> packed u64
// grid: (ROWS_/16, N/512, M_), block: 256 (8 waves, 4 n-tiles each)
// Inner loop is register double-buffered: loads for K-step k+32 are issued
// before the 4 WMMAs of K-step k, so waits overlap matrix-op execution.
__global__ void __launch_bounds__(256) k_gemm_argmax(
    const unsigned short* __restrict__ Xb,
    const unsigned short* __restrict__ Wb,
    unsigned long long* __restrict__ packed,
    int N)
{
  const int rowTile = blockIdx.x;
  const int chunk   = blockIdx.y;
  const int m       = blockIdx.z;
  const int lane = threadIdx.x & 31;
  const int wave = threadIdx.x >> 5;
  const int half = lane >> 4;
  const int l16  = lane & 15;
  const int r0   = rowTile * 16;

  const unsigned short* __restrict__ W = Wb + (size_t)m * N * D_;
  unsigned long long* __restrict__ P   = packed + (size_t)m * ROWS_;

  // A: row (r0+l16) of Xb; half selects the K sub-chunks
  const unsigned short* arow = Xb + (size_t)(r0 + l16) * D_;

  int ncol[4];
  const unsigned short* wrow[4];
#pragma unroll
  for (int j = 0; j < 4; ++j) {
    int nt = chunk * 32 + wave + 8 * j;
    ncol[j] = nt * 16 + l16;                       // this lane's B column
    wrow[j] = W + (size_t)ncol[j] * D_ + 16 * half;
    __builtin_prefetch(wrow[j], 0, 3);
  }

  v8f acc[4];
#pragma unroll
  for (int j = 0; j < 4; ++j) acc[j] = 0;

  auto load_frags = [&](ABFrag& a, ABFrag* b, int k0) {
    a.q[0] = *(const uint4*)(arow + k0 + 16 * half);       // K = 16h..16h+7
    a.q[1] = *(const uint4*)(arow + k0 + 16 * half + 8);   // K = 16h+8..16h+15
#pragma unroll
    for (int j = 0; j < 4; ++j) {
      b[j].q[0] = *(const uint4*)(wrow[j] + k0);           // K = 16h..16h+7
      b[j].q[1] = *(const uint4*)(wrow[j] + k0 + 8);       // K = 16h+8..16h+15
    }
  };
  auto do_wmma = [&](const ABFrag& a, const ABFrag* b) {
#pragma unroll
    for (int j = 0; j < 4; ++j)
      acc[j] = __builtin_amdgcn_wmma_f32_16x16x32_bf16(
          false, a.v, false, b[j].v, (short)0, acc[j], false, false);
  };

  ABFrag a0, b0[4], a1, b1[4];
  load_frags(a0, b0, 0);
  int k0 = 0;
  for (; k0 < D_ - 64; k0 += 64) {
    load_frags(a1, b1, k0 + 32);   // prefetch next K-step into regs
    do_wmma(a0, b0);               // consume current while loads in flight
    load_frags(a0, b0, k0 + 64);
    do_wmma(a1, b1);
  }
  // tail: k0 == D_-64
  load_frags(a1, b1, k0 + 32);
  do_wmma(a0, b0);
  do_wmma(a1, b1);

  // per-row argmax: pack (ordered score, ~col) so max => max score, then min col
  unsigned long long best[8];
#pragma unroll
  for (int i = 0; i < 8; ++i) {
    unsigned long long bb = 0;
#pragma unroll
    for (int j = 0; j < 4; ++j) {
      unsigned long long p = ((unsigned long long)f2ord(acc[j][i]) << 32)
                           | (unsigned long long)(0xFFFFFFFFu - (unsigned)ncol[j]);
      bb = p > bb ? p : bb;
    }
    best[i] = bb;  // row = r0 + 8*half + i (C/D layout)
  }
  // reduce across the 16 lanes of each half (wave32)
#pragma unroll
  for (int s = 1; s < 16; s <<= 1) {
#pragma unroll
    for (int i = 0; i < 8; ++i) {
      unsigned long long o = sxor64(best[i], s);
      best[i] = o > best[i] ? o : best[i];
    }
  }
  __shared__ unsigned long long red[8][16];
  if (l16 == 0) {
#pragma unroll
    for (int i = 0; i < 8; ++i) red[wave][half * 8 + i] = best[i];
  }
  __syncthreads();
  if (threadIdx.x < 16) {
    unsigned long long mx = red[0][threadIdx.x];
#pragma unroll
    for (int w = 1; w < 8; ++w) {
      unsigned long long o = red[w][threadIdx.x];
      mx = o > mx ? o : mx;
    }
    atomicMax(&P[r0 + threadIdx.x], mx);
  }
}

__global__ void k_finalize(const unsigned long long* __restrict__ p,
                           int* __restrict__ codes, int n) {
  int i = blockIdx.x * blockDim.x + threadIdx.x;
  if (i < n) codes[i] = (int)(0xFFFFFFFFu - (unsigned)(p[i] & 0xFFFFFFFFull));
}

// ---------------- suffix automaton build + lookup ----------------
// grid: M_*B_ blocks; thread 0 builds SAM serially, all threads do lookups
__global__ void k_sam(const int* __restrict__ keys, const int* __restrict__ queries,
                      short* __restrict__ nxtAll, int* __restrict__ linkAll,
                      int* __restrict__ lenAll, int* __restrict__ eAll,
                      int* __restrict__ idxA, int* __restrict__ valA)
{
  const int seq = blockIdx.x;                 // m*B_ + b
  const int* kb = keys + (size_t)seq * T_;
  const int* qb = queries + (size_t)seq * T_;
  short* nxt = nxtAll + (size_t)seq * NST_ * KQK_;  // pre-initialized to -1
  int* link = linkAll + (size_t)seq * NST_;
  int* len  = lenAll  + (size_t)seq * NST_;
  int* e    = eAll    + (size_t)seq * NST_;
  __shared__ int s_last;

  if (threadIdx.x == 0) {
    link[0] = -1; len[0] = 0; e[0] = -1;
    int size = 1, last = 0;
    for (int i = 0; i < T_; ++i) {
      int ch = kb[i];
      int cur = size++;
      len[cur] = len[last] + 1;
      e[cur] = i;
      int p = last;
      while (p != -1 && nxt[(size_t)p * KQK_ + ch] == -1) {
        nxt[(size_t)p * KQK_ + ch] = (short)cur;
        p = link[p];
      }
      if (p == -1) {
        link[cur] = 0;
      } else {
        int q = nxt[(size_t)p * KQK_ + ch];
        if (len[p] + 1 == len[q]) {
          link[cur] = q;
        } else {
          int cl = size++;
          len[cl] = len[p] + 1;
          short* dn = nxt + (size_t)cl * KQK_;
          const short* sn = nxt + (size_t)q * KQK_;
          for (int c = 0; c < KQK_; ++c) dn[c] = sn[c];
          link[cl] = link[q];
          e[cl] = e[q];
          while (p != -1 && nxt[(size_t)p * KQK_ + ch] == q) {
            nxt[(size_t)p * KQK_ + ch] = (short)cl;
            p = link[p];
          }
          link[q] = cl;
          link[cur] = cl;
        }
      }
      last = cur;
    }
    s_last = last;
  }
  __syncthreads();
  const int last = s_last;

  for (int t = threadIdx.x; t < T_; t += blockDim.x) {
    int x = qb[t];
    int p = last;
    while (p != -1 && nxt[(size_t)p * KQK_ + x] == -1) p = link[p];
    int q = (p == -1) ? -1 : (int)nxt[(size_t)p * KQK_ + x];
    int* io = idxA + ((size_t)seq * T_ + t) * TOPK_;
    int* vo = valA + ((size_t)seq * T_ + t) * TOPK_;
#pragma unroll
    for (int k = 0; k < TOPK_; ++k) { io[k] = -1; vo[k] = 0; }
    if (q != -1) {
      int rpos = e[q];
      if (rpos >= 0) {
        int n = 0;
        for (int off = 1; off <= TOPK_; ++off) {
          int id = rpos + off;
          if (id >= T_) break;
          int kc = kb[id];
          if (kc != x) { io[n] = id; vo[n] = kc; ++n; }
        }
      }
    }
  }
}

// ---------------- gather-contract output ----------------
// out[b,t,d] = sum_m w_m * sum_k Vemb[m, values[m,b,idx_k], d] * Vemb[m, val_k, d]
// grid: ROWS_ blocks; 256 threads x float4 over D_=1024
__global__ void __launch_bounds__(256) k_route(
    const float* __restrict__ Vemb, const int* __restrict__ values,
    const int* __restrict__ idxA, const int* __restrict__ valA,
    const float* __restrict__ alpha, float* __restrict__ out)
{
  const int bt = blockIdx.x;
  const int b = bt / T_;
  const int tid = threadIdx.x;
  __shared__ int sg[M_][TOPK_];
  __shared__ int sv[M_][TOPK_];
  if (tid < M_ * TOPK_) {
    int m = tid / TOPK_, k = tid % TOPK_;
    size_t o = ((size_t)m * ROWS_ + bt) * TOPK_ + k;
    int id = idxA[o];
    int g = 0, vv = 0;
    if (id >= 0) {                       // masked slots -> row 0 (zeroed)
      g = values[(size_t)m * ROWS_ + (size_t)b * T_ + id];
      vv = valA[o];
    }
    sg[m][k] = g;
    sv[m][k] = vv;
  }
  __syncthreads();

  float w[M_];
  {
    float mx = alpha[0];
    for (int m = 1; m < M_; ++m) mx = fmaxf(mx, alpha[m]);
    float s = 0.f;
    for (int m = 0; m < M_; ++m) { w[m] = __expf(alpha[m] - mx); s += w[m]; }
    for (int m = 0; m < M_; ++m) w[m] /= s;
  }

  const int d0 = tid * 4;
  float4 acc = {0.f, 0.f, 0.f, 0.f};
  for (int m = 0; m < M_; ++m) {
    float4 s = {0.f, 0.f, 0.f, 0.f};
#pragma unroll
    for (int k = 0; k < TOPK_; ++k) {
      const float4 gv = *(const float4*)(Vemb + ((size_t)m * (VV_ + 1) + sg[m][k]) * D_ + d0);
      const float4 lv = *(const float4*)(Vemb + ((size_t)m * (VV_ + 1) + sv[m][k]) * D_ + d0);
      s.x += gv.x * lv.x; s.y += gv.y * lv.y; s.z += gv.z * lv.z; s.w += gv.w * lv.w;
    }
    acc.x += w[m] * s.x; acc.y += w[m] * s.y; acc.z += w[m] * s.z; acc.w += w[m] * s.w;
  }
  *(float4*)(out + (size_t)bt * D_ + d0) = acc;
}

// ---------------- launch ----------------
extern "C" void kernel_launch(void* const* d_in, const int* in_sizes, int n_in,
                              void* d_out, int out_size, void* d_ws, size_t ws_size,
                              hipStream_t stream)
{
  const float* x     = (const float*)d_in[0];
  const float* Wq    = (const float*)d_in[1];
  const float* Wk    = (const float*)d_in[2];
  const float* Wv    = (const float*)d_in[3];
  const float* Vemb  = (const float*)d_in[4];
  const float* alpha = (const float*)d_in[5];
  float* out = (float*)d_out;

  char* base = (char*)d_ws;
  size_t off = 0;
  auto alloc = [&](size_t bytes) -> void* {
    off = (off + 255) & ~(size_t)255;
    void* p = base + off;
    off += bytes;
    return p;
  };
  unsigned short* xb  = (unsigned short*)alloc((size_t)ROWS_ * D_ * 2);
  unsigned short* wqb = (unsigned short*)alloc((size_t)M_ * KQK_ * D_ * 2);
  unsigned short* wkb = (unsigned short*)alloc((size_t)M_ * KQK_ * D_ * 2);
  unsigned short* wvb = (unsigned short*)alloc((size_t)M_ * VV_ * D_ * 2);
  unsigned long long* pq = (unsigned long long*)alloc((size_t)M_ * ROWS_ * 8);
  unsigned long long* pk = (unsigned long long*)alloc((size_t)M_ * ROWS_ * 8);
  unsigned long long* pv = (unsigned long long*)alloc((size_t)M_ * ROWS_ * 8);
  int* qcodes = (int*)alloc((size_t)M_ * ROWS_ * 4);
  int* kcodes = (int*)alloc((size_t)M_ * ROWS_ * 4);
  int* vcodes = (int*)alloc((size_t)M_ * ROWS_ * 4);
  short* nxt = (short*)alloc((size_t)M_ * B_ * NST_ * KQK_ * 2);
  int* link = (int*)alloc((size_t)M_ * B_ * NST_ * 4);
  int* len  = (int*)alloc((size_t)M_ * B_ * NST_ * 4);
  int* e    = (int*)alloc((size_t)M_ * B_ * NST_ * 4);
  int* idxA = (int*)alloc((size_t)M_ * ROWS_ * TOPK_ * 4);
  int* valA = (int*)alloc((size_t)M_ * ROWS_ * TOPK_ * 4);

  // fp32 -> bf16 for WMMA operands
  k_cvt_bf16<<<4096, 256, 0, stream>>>(x,  xb,  (size_t)ROWS_ * D_ / 4);
  k_cvt_bf16<<<2048, 256, 0, stream>>>(Wq, wqb, (size_t)M_ * KQK_ * D_ / 4);
  k_cvt_bf16<<<2048, 256, 0, stream>>>(Wk, wkb, (size_t)M_ * KQK_ * D_ / 4);
  k_cvt_bf16<<<4096, 256, 0, stream>>>(Wv, wvb, (size_t)M_ * VV_ * D_ / 4);

  // zero packed argmax accumulators; -1-fill SAM transition tables
  k_fill_u32<<<64, 256, 0, stream>>>((unsigned*)pq, (size_t)M_ * ROWS_ * 2, 0u);
  k_fill_u32<<<64, 256, 0, stream>>>((unsigned*)pk, (size_t)M_ * ROWS_ * 2, 0u);
  k_fill_u32<<<64, 256, 0, stream>>>((unsigned*)pv, (size_t)M_ * ROWS_ * 2, 0u);
  k_fill_u32<<<4096, 256, 0, stream>>>((unsigned*)nxt,
      (size_t)M_ * B_ * NST_ * KQK_ / 2, 0xFFFFFFFFu);

  // WMMA GEMM + argmax (queries, keys, values)
  k_gemm_argmax<<<dim3(ROWS_ / 16, KQK_ / 512, M_), 256, 0, stream>>>(xb, wqb, pq, KQK_);
  k_gemm_argmax<<<dim3(ROWS_ / 16, KQK_ / 512, M_), 256, 0, stream>>>(xb, wkb, pk, KQK_);
  k_gemm_argmax<<<dim3(ROWS_ / 16, VV_  / 512, M_), 256, 0, stream>>>(xb, wvb, pv, VV_);

  k_finalize<<<(M_ * ROWS_ + 255) / 256, 256, 0, stream>>>(pq, qcodes, M_ * ROWS_);
  k_finalize<<<(M_ * ROWS_ + 255) / 256, 256, 0, stream>>>(pk, kcodes, M_ * ROWS_);
  k_finalize<<<(M_ * ROWS_ + 255) / 256, 256, 0, stream>>>(pv, vcodes, M_ * ROWS_);

  // suffix automaton build + topk lookups
  k_sam<<<M_ * B_, 256, 0, stream>>>(kcodes, qcodes, nxt, link, len, e, idxA, valA);

  // gather-contract + softmax(alpha) route mix
  k_route<<<ROWS_, 256, 0, stream>>>(Vemb, vcodes, idxA, valA, alpha, out);

  (void)in_sizes; (void)n_in; (void)out_size; (void)ws_size;
}
